// Attn_9715216024104
// MI455X (gfx1250) — matrix operand order repaired and checked by reading, not compiled, not run on hardware
//
#include <hip/hip_runtime.h>
#include <hip/hip_bf16.h>
#include <math.h>
#include <stdint.h>

typedef __attribute__((ext_vector_type(16))) _Float16 v16h;
typedef __attribute__((ext_vector_type(8)))  float    v8f;

#define B_    2
#define T_    2048
#define C_    256
#define H_    16
#define HD_   96
#define VD_   32
#define KEEP_ 512
#define WIN_  128
#define BT_   (B_*T_)

// ---------------------------------------------------------------------------
// f32 -> f16 conversion
// ---------------------------------------------------------------------------
__global__ void cvt_f32_f16(const float* __restrict__ s, _Float16* __restrict__ d, int n) {
  int i = blockIdx.x * blockDim.x + threadIdx.x;
  if (i < n) d[i] = (_Float16)s[i];
}

__global__ void zero_f32(float* __restrict__ p, int n) {
  int i = blockIdx.x * blockDim.x + threadIdx.x;
  if (i < n) p[i] = 0.0f;
}

// ---------------------------------------------------------------------------
// Generic WMMA GEMM: C[M,N] = A[M,K] * B[K,N], f16 in, f32 accumulate.
// One wave32 per 16x16 output tile; K stepped by 32 (all K here are %32==0).
// A-frag layout (16-bit A 16x32): lane<16 -> K {0..15}, lane>=16 -> {16..31} (contiguous k, as B)
// B-frag layout (16-bit B 32x16): lane&15 = N, lane>>4 selects K-half, elem e = K offset
// C layout: elem r -> row = r + 8*(lane>>4), col = lane&15
// ---------------------------------------------------------------------------
__global__ __launch_bounds__(256)
void gemm16(const _Float16* __restrict__ A, const _Float16* __restrict__ Bm,
            void* __restrict__ Cm, int M, int N, int K, int cf32, int nwaves) {
  int gw = (blockIdx.x * blockDim.x + threadIdx.x) >> 5;
  if (gw >= nwaves) return;
  int lane = threadIdx.x & 31;
  int ntiles = N >> 4;
  int mt = gw / ntiles, nt = gw - mt * ntiles;
  int m_a = lane & 15, kba = (lane >> 4) << 4;
  int n_b = lane & 15, kbb = (lane >> 4) << 4;
  const _Float16* Ar = A + (size_t)(mt * 16 + m_a) * K;
  v8f acc = {};
  for (int k0 = 0; k0 < K; k0 += 32) {
    if (k0 + 32 < K) {  // hide latency of next K-chunk (-> global_prefetch_b8)
      __builtin_prefetch(Ar + k0 + 32, 0, 0);
      __builtin_prefetch(Bm + (size_t)(k0 + 32 + kbb) * N + nt * 16 + n_b, 0, 0);
    }
    v16h a, b;
#pragma unroll
    for (int i = 0; i < 8; ++i) {
      a[i]     = Ar[k0 + kba + i];
      a[i + 8] = Ar[k0 + kba + 8 + i];
    }
    const _Float16* Br = Bm + (size_t)(k0 + kbb) * N + nt * 16 + n_b;
#pragma unroll
    for (int e = 0; e < 16; ++e) b[e] = Br[(size_t)e * N];
    acc = __builtin_amdgcn_wmma_f32_16x16x32_f16(false, a, false, b, (short)0, acc, false, false);
  }
  int gbase = (lane >> 4) << 3;
  int col = nt * 16 + n_b;
  if (cf32) {
    float* C = (float*)Cm;
#pragma unroll
    for (int r = 0; r < 8; ++r) C[(size_t)(mt * 16 + gbase + r) * N + col] = acc[r];
  } else {
    _Float16* C = (_Float16*)Cm;
#pragma unroll
    for (int r = 0; r < 8; ++r) C[(size_t)(mt * 16 + gbase + r) * N + col] = (_Float16)acc[r];
  }
}

// ---------------------------------------------------------------------------
// RMS norm over rows, in place (f16 buffer, f32 math).
// ---------------------------------------------------------------------------
__global__ void rms_rows(_Float16* __restrict__ buf, const float* __restrict__ w,
                         int rows, int dim) {
  int r = blockIdx.x * blockDim.x + threadIdx.x;
  if (r >= rows) return;
  _Float16* p = buf + (size_t)r * dim;
  float ss = 0.f;
  for (int i = 0; i < dim; ++i) { float v = (float)p[i]; ss += v * v; }
  float inv = rsqrtf(ss / (float)dim + 1e-6f);
  for (int i = 0; i < dim; ++i) p[i] = (_Float16)((float)p[i] * inv * w[i]);
}

// ---------------------------------------------------------------------------
// scores = x @ W_imp ; gate logits = x @ Wgate   (tiny N, scalar dot products)
// ---------------------------------------------------------------------------
__global__ void scores_gate(const float* __restrict__ x, const float* __restrict__ Wimp,
                            const float* __restrict__ Wgate, float* __restrict__ scores,
                            float* __restrict__ glog) {
  int i = blockIdx.x * blockDim.x + threadIdx.x;
  if (i >= BT_) return;
  const float* xr = x + (size_t)i * C_;
  float s = 0.f, g0 = 0.f, g1 = 0.f, g2 = 0.f;
  for (int c = 0; c < C_; ++c) {
    float xv = xr[c];
    s  += xv * Wimp[c];
    g0 += xv * Wgate[c * 3 + 0];
    g1 += xv * Wgate[c * 3 + 1];
    g2 += xv * Wgate[c * 3 + 2];
  }
  scores[i] = s;
  glog[(size_t)i * 3 + 0] = g0;
  glog[(size_t)i * 3 + 1] = g1;
  glog[(size_t)i * 3 + 2] = g2;
}

// gate[b] = softmax(mean_t(glog[b,t,:]))   (deterministic tree reduce, 1 block/batch)
__global__ void gate_reduce(const float* __restrict__ glog, float* __restrict__ gates) {
  int b = blockIdx.x, tid = threadIdx.x;
  __shared__ float s0[256], s1[256], s2[256];
  float a0 = 0.f, a1 = 0.f, a2 = 0.f;
  for (int t = tid; t < T_; t += 256) {
    const float* p = glog + ((size_t)b * T_ + t) * 3;
    a0 += p[0]; a1 += p[1]; a2 += p[2];
  }
  s0[tid] = a0; s1[tid] = a1; s2[tid] = a2;
  __syncthreads();
  for (int s = 128; s > 0; s >>= 1) {
    if (tid < s) { s0[tid] += s0[tid + s]; s1[tid] += s1[tid + s]; s2[tid] += s2[tid + s]; }
    __syncthreads();
  }
  if (tid == 0) {
    float m0 = s0[0] / T_, m1 = s1[0] / T_, m2 = s2[0] / T_;
    float mx = fmaxf(m0, fmaxf(m1, m2));
    float e0 = expf(m0 - mx), e1 = expf(m1 - mx), e2 = expf(m2 - mx);
    float sum = e0 + e1 + e2;
    gates[b * 3 + 0] = e0 / sum;
    gates[b * 3 + 1] = e1 / sum;
    gates[b * 3 + 2] = e2 / sum;
  }
}

// ---------------------------------------------------------------------------
// top-k by rank counting (ties -> lower index wins, matching jax top_k),
// compaction in ascending index order == sort(top_k indices).
// ---------------------------------------------------------------------------
__global__ void topk_sel(const float* __restrict__ scores, int* __restrict__ flags,
                         int* __restrict__ idx) {
  int b = blockIdx.x, tid = threadIdx.x;
  const float* s = scores + (size_t)b * T_;
  int* f = flags + (size_t)b * T_;
  for (int i = tid; i < T_; i += 256) {
    float si = s[i];
    int rank = 0;
    for (int j = 0; j < T_; ++j) {
      float sj = s[j];
      rank += (sj > si) || (sj == si && j < i);
    }
    f[i] = (rank < KEEP_) ? 1 : 0;
  }
  __syncthreads();
  if (tid == 0) {
    int pos = 0;
    for (int i = 0; i < T_ && pos < KEEP_; ++i)
      if (f[i]) idx[b * KEEP_ + pos++] = i;
  }
}

__global__ void gather_sel(const float* __restrict__ x, const int* __restrict__ idx,
                           _Float16* __restrict__ Xsel) {
  int i = blockIdx.x * blockDim.x + threadIdx.x;
  if (i >= B_ * KEEP_ * C_) return;
  int c = i % C_;
  int r = i / C_;
  int j = r % KEEP_;
  int b = r / KEEP_;
  int src = idx[b * KEEP_ + j];
  Xsel[i] = (_Float16)x[((size_t)b * T_ + src) * C_ + c];
}

// ---------------------------------------------------------------------------
// RoPE helper: dim=64, freqs f_j = 10000^(-j/32), j = 0..31
// ---------------------------------------------------------------------------
__device__ inline void rope_cs(int pos, int j, float& c, float& s) {
  float ang = (float)pos * __expf(-(float)j * 0.28782313662425575f); // ln(1e4)/32
  c = cosf(ang); s = sinf(ang);
}

// Build head-major f16 tensors with RoPE applied: q, k1 (kn + kr/16 roped), v1, kw, vw.
__global__ void build_heads(const _Float16* __restrict__ qn, const _Float16* __restrict__ qr,
                            const _Float16* __restrict__ kn, const _Float16* __restrict__ kr,
                            const _Float16* __restrict__ v1r, const _Float16* __restrict__ kwr,
                            const _Float16* __restrict__ vwr,
                            _Float16* __restrict__ q, _Float16* __restrict__ k1,
                            _Float16* __restrict__ v1, _Float16* __restrict__ kw,
                            _Float16* __restrict__ vw) {
  int i = blockIdx.x * blockDim.x + threadIdx.x;
  if (i >= B_ * H_ * T_) return;
  int t  = i % T_;
  int bh = i / T_;
  int h  = bh & 15;
  int b  = bh >> 4;
  size_t row = (size_t)b * T_ + t;

  _Float16* qd = q + (size_t)i * HD_;
  const _Float16* qnp = qn + row * (H_ * 32) + h * 32;
  for (int d = 0; d < 32; ++d) qd[d] = qnp[d];
  const _Float16* qrp = qr + row * (H_ * 64) + (size_t)h * 64;
  for (int j = 0; j < 32; ++j) {
    float c, s; rope_cs(t, j, c, s);
    float xr = (float)qrp[j], xi = (float)qrp[32 + j];
    qd[32 + j] = (_Float16)(xr * c - xi * s);
    qd[64 + j] = (_Float16)(xr * s + xi * c);
  }

  _Float16* kd = k1 + (size_t)i * HD_;
  const _Float16* knp = kn + row * (H_ * 32) + h * 32;
  for (int d = 0; d < 32; ++d) kd[d] = knp[d];
  const _Float16* krp = kr + row * 64;
  for (int j = 0; j < 32; ++j) {
    float c, s; rope_cs(t, j, c, s);
    float xr = (float)krp[j] * 0.0625f, xi = (float)krp[32 + j] * 0.0625f; // /N_HEAD
    kd[32 + j] = (_Float16)(xr * c - xi * s);
    kd[64 + j] = (_Float16)(xr * s + xi * c);
  }

  _Float16* vd = v1 + (size_t)i * VD_;
  const _Float16* vp = v1r + row * (H_ * 32) + h * 32;
  for (int d = 0; d < 32; ++d) vd[d] = vp[d];

  _Float16* kwd = kw + (size_t)i * HD_;
  const _Float16* kwp = kwr + row * (H_ * HD_) + (size_t)h * HD_;
  for (int d = 0; d < 32; ++d) kwd[d] = kwp[d];
  for (int j = 0; j < 32; ++j) {
    float c, s; rope_cs(t, j, c, s);
    float xr = (float)kwp[32 + j], xi = (float)kwp[64 + j];
    kwd[32 + j] = (_Float16)(xr * c - xi * s);
    kwd[64 + j] = (_Float16)(xr * s + xi * c);
  }

  _Float16* vwd = vw + (size_t)i * VD_;
  const _Float16* vwp = vwr + row * (H_ * 32) + h * 32;
  for (int d = 0; d < 32; ++d) vwd[d] = vwp[d];
}

// Build head-major ks/vs; RoPE position = compacted index j (as in reference).
__global__ void build_sel(const _Float16* __restrict__ ksr, const _Float16* __restrict__ vsr,
                          _Float16* __restrict__ ks, _Float16* __restrict__ vs) {
  int i = blockIdx.x * blockDim.x + threadIdx.x;
  if (i >= B_ * H_ * KEEP_) return;
  int j  = i % KEEP_;
  int bh = i / KEEP_;
  int h  = bh & 15;
  int b  = bh >> 4;
  size_t row = (size_t)b * KEEP_ + j;

  _Float16* kd = ks + (size_t)i * HD_;
  const _Float16* kp = ksr + row * (H_ * HD_) + (size_t)h * HD_;
  for (int d = 0; d < 32; ++d) kd[d] = kp[d];
  for (int jj = 0; jj < 32; ++jj) {
    float c, s; rope_cs(j, jj, c, s);
    float xr = (float)kp[32 + jj], xi = (float)kp[64 + jj];
    kd[32 + jj] = (_Float16)(xr * c - xi * s);
    kd[64 + jj] = (_Float16)(xr * s + xi * c);
  }
  _Float16* vd = vs + (size_t)i * VD_;
  const _Float16* vp = vsr + row * (H_ * 32) + h * 32;
  for (int d = 0; d < 32; ++d) vd[d] = vp[d];
}

// ---------------------------------------------------------------------------
// Flash-attention with WMMA + async double-buffered K/V staging.
// One wave32 per 16-row Q tile per (b,h); 4 waves per block, each with a
// 16KB LDS slab: K slots [0,6144)x2, V slots [12288,14336)x2 bytes.
// Per 32-key block: stage K (12x async b128) + V (4x async b128) into the
// non-current slot while WMMAs consume the current one (ASYNCcnt pipelining).
// P (16x32 f16, 1KB) overlays the *current* K slot after S-WMMAs consume it.
// mode: 0 = causal full, 1 = no mask (selected), 2 = causal window.
// Gating folded in: Oacc += gates[b,mode] * O/L.
// ---------------------------------------------------------------------------
__global__ __launch_bounds__(128)
void attn_wmma(const _Float16* __restrict__ Q, const _Float16* __restrict__ K,
               const _Float16* __restrict__ V, const float* __restrict__ gates,
               float* __restrict__ Oacc, int Tk, int mode, int nwaves) {
  __shared__ _Float16 slab[4 * 8192];  // 64 KB total, 16 KB per wave
  int wib = threadIdx.x >> 5;
  int gw = blockIdx.x * 4 + wib;
  if (gw >= nwaves) return;
  int lane = threadIdx.x & 31;
  const int qtiles = T_ / 16;
  int bh = gw / qtiles;
  int qt = gw - bh * qtiles;
  int b  = bh >> 4;
  int q0 = qt * 16;

  const _Float16* Qb = Q + (size_t)bh * T_ * HD_;
  const _Float16* Kb = K + (size_t)bh * Tk * HD_;
  const _Float16* Vb = V + (size_t)bh * Tk * VD_;
  _Float16* ws0 = slab + wib * 8192;
  uint32_t slabByte = (uint32_t)(uintptr_t)(void*)ws0;

  int m_a = lane & 15, kba = (lane >> 4) << 4;
  int n_b = lane & 15, kbb = (lane >> 4) << 4;
  int gbase = (lane >> 4) << 3;

  // cached Q A-fragments (3 chunks of K=32 over HD=96)
  v16h aq[3];
  {
    const _Float16* qr = Qb + (size_t)(q0 + m_a) * HD_;
#pragma unroll
    for (int c = 0; c < 3; ++c)
#pragma unroll
      for (int i = 0; i < 8; ++i) {
        aq[c][i]     = qr[c * 32 + kba + i];
        aq[c][i + 8] = qr[c * 32 + kba + 8 + i];
      }
  }

  float mrow[8], lrow[8];
#pragma unroll
  for (int r = 0; r < 8; ++r) { mrow[r] = -3.0e38f; lrow[r] = 0.f; }
  v8f O0 = {}, O1 = {};

  int kb0 = 0, kb1 = (Tk >> 5) - 1;
  if (mode == 0) kb1 = (q0 + 15) >> 5;
  else if (mode == 2) {
    kb1 = (q0 + 15) >> 5;
    int lo = q0 - (WIN_ - 1);
    if (lo < 0) lo = 0;
    kb0 = lo >> 5;
  }

  // async stage of one 32-key K/V tile into LDS slot (16x 512B b128 transfers)
  auto stage = [&](int kb, int slot) {
    uint64_t kg = (uint64_t)(uintptr_t)(Kb + (size_t)kb * 32 * HD_) + (uint32_t)(lane * 16);
    uint64_t vg = (uint64_t)(uintptr_t)(Vb + (size_t)kb * 32 * VD_) + (uint32_t)(lane * 16);
    uint32_t kl = slabByte + slot * 6144 + lane * 16;
    uint32_t vl = slabByte + 12288 + slot * 2048 + lane * 16;
#pragma unroll
    for (int i = 0; i < 12; ++i)
      asm volatile("global_load_async_to_lds_b128 %0, %1, off"
                   :: "v"(kl + i * 512), "v"(kg + (uint32_t)(i * 512)) : "memory");
#pragma unroll
    for (int i = 0; i < 4; ++i)
      asm volatile("global_load_async_to_lds_b128 %0, %1, off"
                   :: "v"(vl + i * 512), "v"(vg + (uint32_t)(i * 512)) : "memory");
  };

  const float scale = 0.10206207261596576f; // 1/sqrt(HD)

  int cur = 0;
  stage(kb0, 0);
  for (int kb = kb0; kb <= kb1; ++kb) {
    if (kb < kb1) {
      stage(kb + 1, cur ^ 1);                               // prefetch next tile
      asm volatile("s_wait_asynccnt 0x10" ::: "memory");    // current tile resident
    } else {
      asm volatile("s_wait_asynccnt 0x0" ::: "memory");
    }
    const _Float16* Kc = ws0 + cur * 3072;          // 32 x 96 f16 tile
    const _Float16* Vc = ws0 + 6144 + cur * 1024;   // 32 x 32 f16 tile

    v8f s0 = {}, s1 = {};
#pragma unroll
    for (int c = 0; c < 3; ++c) {
      v16h b0, b1;
      const _Float16* k0p = Kc + (size_t)n_b * HD_ + c * 32 + kbb;
      const _Float16* k1p = Kc + (size_t)(16 + n_b) * HD_ + c * 32 + kbb;
#pragma unroll
      for (int e = 0; e < 16; ++e) { b0[e] = k0p[e]; b1[e] = k1p[e]; }
      s0 = __builtin_amdgcn_wmma_f32_16x16x32_f16(false, aq[c], false, b0, (short)0, s0, false, false);
      s1 = __builtin_amdgcn_wmma_f32_16x16x32_f16(false, aq[c], false, b1, (short)0, s1, false, false);
    }

    _Float16* pp = ws0 + cur * 3072;  // P overlays current (consumed) K slot
#pragma unroll
    for (int r = 0; r < 8; ++r) {
      float v0 = s0[r] * scale, v1 = s1[r] * scale;
      int qg  = q0 + gbase + r;
      int kg0 = kb * 32 + n_b, kg1 = kg0 + 16;
      if (mode == 0) {
        if (kg0 > qg) v0 = -1e9f;
        if (kg1 > qg) v1 = -1e9f;
      } else if (mode == 2) {
        if (kg0 > qg || (qg - kg0) >= WIN_) v0 = -1e9f;
        if (kg1 > qg || (qg - kg1) >= WIN_) v1 = -1e9f;
      }
      float lm = fmaxf(v0, v1);
#pragma unroll
      for (int off = 8; off >= 1; off >>= 1) lm = fmaxf(lm, __shfl_xor(lm, off, 32));
      float nm = fmaxf(mrow[r], lm);
      float alpha = __expf(mrow[r] - nm);
      mrow[r] = nm;
      float p0 = __expf(v0 - nm), p1 = __expf(v1 - nm);
      float ls = p0 + p1;
#pragma unroll
      for (int off = 8; off >= 1; off >>= 1) ls += __shfl_xor(ls, off, 32);
      lrow[r] = lrow[r] * alpha + ls;
      O0[r] *= alpha;
      O1[r] *= alpha;
      pp[(gbase + r) * 32 + n_b]      = (_Float16)p0;
      pp[(gbase + r) * 32 + 16 + n_b] = (_Float16)p1;
    }
    // wave-local LDS write -> cross-lane read ordering (CDNA5 split counter)
    asm volatile("s_wait_dscnt 0x0" ::: "memory");
    v16h ap, bv0, bv1;
#pragma unroll
    for (int i = 0; i < 8; ++i) {
      ap[i]     = pp[m_a * 32 + kba + i];
      ap[i + 8] = pp[m_a * 32 + kba + 8 + i];
    }
#pragma unroll
    for (int e = 0; e < 16; ++e) {
      bv0[e] = Vc[(size_t)(kbb + e) * VD_ + n_b];
      bv1[e] = Vc[(size_t)(kbb + e) * VD_ + 16 + n_b];
    }
    O0 = __builtin_amdgcn_wmma_f32_16x16x32_f16(false, ap, false, bv0, (short)0, O0, false, false);
    O1 = __builtin_amdgcn_wmma_f32_16x16x32_f16(false, ap, false, bv1, (short)0, O1, false, false);
    cur ^= 1;
  }

  float g = gates[b * 3 + mode];
#pragma unroll
  for (int r = 0; r < 8; ++r) {
    float inv = g / fmaxf(lrow[r], 1e-20f);
    size_t o = ((size_t)bh * T_ + (q0 + gbase + r)) * VD_ + n_b;
    Oacc[o]      += O0[r] * inv;
    Oacc[o + 16] += O1[r] * inv;
  }
}

// (B,H,T,V) f32 -> (B,T,H*V) f16 for the final projection GEMM
__global__ void combine(const float* __restrict__ oacc, _Float16* __restrict__ ocomb) {
  int i = blockIdx.x * blockDim.x + threadIdx.x;
  if (i >= BT_ * (H_ * VD_)) return;
  int v  = i & 31;
  int h  = (i >> 5) & 15;
  int bt = i >> 9;
  int t  = bt % T_;
  int b  = bt / T_;
  ocomb[i] = (_Float16)oacc[(((size_t)b * H_ + h) * T_ + t) * VD_ + v];
}

// ---------------------------------------------------------------------------
// Host side
// ---------------------------------------------------------------------------
static inline void launch_gemm(const _Float16* A, const _Float16* B, void* C,
                               int M, int N, int K, int cf32, hipStream_t st) {
  int waves = (M / 16) * (N / 16);
  int blocks = (waves * 32 + 255) / 256;
  gemm16<<<blocks, 256, 0, st>>>(A, B, C, M, N, K, cf32, waves);
}

static inline void launch_cvt(const float* s, _Float16* d, int n, hipStream_t st) {
  cvt_f32_f16<<<(n + 255) / 256, 256, 0, st>>>(s, d, n);
}

extern "C" void kernel_launch(void* const* d_in, const int* in_sizes, int n_in,
                              void* d_out, int out_size, void* d_ws, size_t ws_size,
                              hipStream_t stream) {
  const float* x       = (const float*)d_in[0];
  const float* Wcq     = (const float*)d_in[1];
  const float* qnw     = (const float*)d_in[2];
  const float* Wq_nope = (const float*)d_in[3];
  const float* Wq_rope = (const float*)d_in[4];
  const float* Wckv    = (const float*)d_in[5];
  const float* kvnw    = (const float*)d_in[6];
  const float* Wk_nope = (const float*)d_in[7];
  const float* Wv      = (const float*)d_in[8];
  const float* Wk_rope = (const float*)d_in[9];
  const float* W_imp   = (const float*)d_in[10];
  const float* Wsel_k  = (const float*)d_in[11];
  const float* Wsel_v  = (const float*)d_in[12];
  const float* Wwin_k  = (const float*)d_in[13];
  const float* Wwin_v  = (const float*)d_in[14];
  const float* Wgate   = (const float*)d_in[15];
  const float* Wproj   = (const float*)d_in[16];

  char* p = (char*)d_ws;
  auto alloc = [&](size_t bytes) -> void* {
    void* r = (void*)p;
    p += (bytes + 255) & ~(size_t)255;
    return r;
  };
  auto hbuf = [&](size_t elems) -> _Float16* { return (_Float16*)alloc(elems * 2); };
  auto fbuf = [&](size_t elems) -> float*    { return (float*)alloc(elems * 4); };
  auto ibuf = [&](size_t elems) -> int*      { return (int*)alloc(elems * 4); };

  // f16 copies of x + weights
  _Float16* Xh  = hbuf((size_t)BT_ * C_);
  _Float16* Wcq_h  = hbuf(256 * 96);
  _Float16* Wqn_h  = hbuf(96 * 512);
  _Float16* Wqr_h  = hbuf(96 * 1024);
  _Float16* Wckv_h = hbuf(256 * 32);
  _Float16* Wkn_h  = hbuf(32 * 512);
  _Float16* Wv_h   = hbuf(32 * 512);
  _Float16* Wkr_h  = hbuf(256 * 64);
  _Float16* Wsk_h  = hbuf(256 * 1536);
  _Float16* Wsv_h  = hbuf(256 * 512);
  _Float16* Wwk_h  = hbuf(256 * 1536);
  _Float16* Wwv_h  = hbuf(256 * 512);
  _Float16* Wpj_h  = hbuf(512 * 256);

  // projection outputs (f16, raw token-major)
  _Float16* cq_h   = hbuf((size_t)BT_ * 96);
  _Float16* ckv_h  = hbuf((size_t)BT_ * 32);
  _Float16* qn_r   = hbuf((size_t)BT_ * 512);
  _Float16* qr_r   = hbuf((size_t)BT_ * 1024);
  _Float16* kn_r   = hbuf((size_t)BT_ * 512);
  _Float16* v1_r   = hbuf((size_t)BT_ * 512);
  _Float16* kr_r   = hbuf((size_t)BT_ * 64);
  _Float16* kw_r   = hbuf((size_t)BT_ * 1536);
  _Float16* vw_r   = hbuf((size_t)BT_ * 512);
  _Float16* Xsel_h = hbuf((size_t)B_ * KEEP_ * C_);
  _Float16* ks_r   = hbuf((size_t)B_ * KEEP_ * 1536);
  _Float16* vs_r   = hbuf((size_t)B_ * KEEP_ * 512);

  // head-major f16 tensors
  _Float16* q_h  = hbuf((size_t)B_ * H_ * T_ * HD_);
  _Float16* k1_h = hbuf((size_t)B_ * H_ * T_ * HD_);
  _Float16* v1_h = hbuf((size_t)B_ * H_ * T_ * VD_);
  _Float16* kw_h = hbuf((size_t)B_ * H_ * T_ * HD_);
  _Float16* vw_h = hbuf((size_t)B_ * H_ * T_ * VD_);
  _Float16* ks_h = hbuf((size_t)B_ * H_ * KEEP_ * HD_);
  _Float16* vs_h = hbuf((size_t)B_ * H_ * KEEP_ * VD_);
  _Float16* ocomb = hbuf((size_t)BT_ * H_ * VD_);

  float* scores = fbuf(BT_);
  float* glog   = fbuf((size_t)BT_ * 3);
  float* gates  = fbuf(B_ * 3);
  float* oacc   = fbuf((size_t)B_ * H_ * T_ * VD_);
  int*   flags  = ibuf((size_t)B_ * T_);
  int*   idx    = ibuf((size_t)B_ * KEEP_);

  // --- stage 0: f32 -> f16 conversions
  launch_cvt(x, Xh, BT_ * C_, stream);
  launch_cvt(Wcq, Wcq_h, 256 * 96, stream);
  launch_cvt(Wq_nope, Wqn_h, 96 * 512, stream);
  launch_cvt(Wq_rope, Wqr_h, 96 * 1024, stream);
  launch_cvt(Wckv, Wckv_h, 256 * 32, stream);
  launch_cvt(Wk_nope, Wkn_h, 32 * 512, stream);
  launch_cvt(Wv, Wv_h, 32 * 512, stream);
  launch_cvt(Wk_rope, Wkr_h, 256 * 64, stream);
  launch_cvt(Wsel_k, Wsk_h, 256 * 1536, stream);
  launch_cvt(Wsel_v, Wsv_h, 256 * 512, stream);
  launch_cvt(Wwin_k, Wwk_h, 256 * 1536, stream);
  launch_cvt(Wwin_v, Wwv_h, 256 * 512, stream);
  launch_cvt(Wproj, Wpj_h, 512 * 256, stream);

  // --- stage 1: latent projections + RMS
  launch_gemm(Xh, Wcq_h, cq_h, BT_, 96, 256, 0, stream);
  rms_rows<<<(BT_ + 255) / 256, 256, 0, stream>>>(cq_h, qnw, BT_, 96);
  launch_gemm(Xh, Wckv_h, ckv_h, BT_, 32, 256, 0, stream);
  rms_rows<<<(BT_ + 255) / 256, 256, 0, stream>>>(ckv_h, kvnw, BT_, 32);

  // --- stage 2: Q/K/V projections
  launch_gemm(cq_h, Wqn_h, qn_r, BT_, 512, 96, 0, stream);
  launch_gemm(cq_h, Wqr_h, qr_r, BT_, 1024, 96, 0, stream);
  launch_gemm(ckv_h, Wkn_h, kn_r, BT_, 512, 32, 0, stream);
  launch_gemm(ckv_h, Wv_h, v1_r, BT_, 512, 32, 0, stream);
  launch_gemm(Xh, Wkr_h, kr_r, BT_, 64, 256, 0, stream);
  launch_gemm(Xh, Wwk_h, kw_r, BT_, 1536, 256, 0, stream);
  launch_gemm(Xh, Wwv_h, vw_r, BT_, 512, 256, 0, stream);

  // --- stage 3: gate + importance top-k + selected projections
  scores_gate<<<(BT_ + 255) / 256, 256, 0, stream>>>(x, W_imp, Wgate, scores, glog);
  gate_reduce<<<B_, 256, 0, stream>>>(glog, gates);
  topk_sel<<<B_, 256, 0, stream>>>(scores, flags, idx);
  {
    int n = B_ * KEEP_ * C_;
    gather_sel<<<(n + 255) / 256, 256, 0, stream>>>(x, idx, Xsel_h);
  }
  launch_gemm(Xsel_h, Wsk_h, ks_r, B_ * KEEP_, 1536, 256, 0, stream);
  launch_gemm(Xsel_h, Wsv_h, vs_r, B_ * KEEP_, 512, 256, 0, stream);

  // --- stage 4: head-major layout + RoPE
  {
    int n = B_ * H_ * T_;
    build_heads<<<(n + 255) / 256, 256, 0, stream>>>(qn_r, qr_r, kn_r, kr_r, v1_r, kw_r, vw_r,
                                                     q_h, k1_h, v1_h, kw_h, vw_h);
    int m = B_ * H_ * KEEP_;
    build_sel<<<(m + 255) / 256, 256, 0, stream>>>(ks_r, vs_r, ks_h, vs_h);
  }

  // --- stage 5: three attention branches, gated accumulation
  {
    int n = B_ * H_ * T_ * VD_;
    zero_f32<<<(n + 255) / 256, 256, 0, stream>>>(oacc, n);
    int nwaves = B_ * H_ * (T_ / 16);
    int blocks = (nwaves + 3) / 4;
    attn_wmma<<<blocks, 128, 0, stream>>>(q_h, k1_h, v1_h, gates, oacc, T_, 0, nwaves);
    attn_wmma<<<blocks, 128, 0, stream>>>(q_h, ks_h, vs_h, gates, oacc, KEEP_, 1, nwaves);
    attn_wmma<<<blocks, 128, 0, stream>>>(q_h, kw_h, vw_h, gates, oacc, T_, 2, nwaves);
  }

  // --- stage 6: combine heads + final projection (f32 out)
  {
    int n = BT_ * H_ * VD_;
    combine<<<(n + 255) / 256, 256, 0, stream>>>(oacc, ocomb);
  }
  launch_gemm(ocomb, Wpj_h, d_out, BT_, 256, 512, 1, stream);

  (void)in_sizes; (void)n_in; (void)out_size; (void)ws_size;
}
